// RelationEncoder_15006615732503
// MI455X (gfx1250) — matrix operand order repaired and checked by reading, not compiled, not run on hardware
//
#include <hip/hip_runtime.h>
#include <hip/hip_bf16.h>

#define BV 32
#define NN 50
#define M1 (BV*NN)        // 1600 node rows
#define PP (BV*NN*NN)     // 80000 pair rows
#define DIN 2048
#define DD 512
#define DP 608            // 592 padded to mult of 32
#define THRV 0.5f

typedef _Float16 v8h_t  __attribute__((ext_vector_type(8)));
typedef _Float16 v16h_t __attribute__((ext_vector_type(16)));
typedef float    v8f_t  __attribute__((ext_vector_type(8)));

// ---------------- mask + f16 convert of images, write ~mask output -------------
__global__ void k_mask_img(const float* __restrict__ img, _Float16* __restrict__ a_h,
                           float* __restrict__ maskf, float* __restrict__ am_out) {
    int r = blockIdx.x, t = threadIdx.x;
    const float* row = img + (size_t)r * DIN;
    _Float16* orow = a_h + (size_t)r * DIN;
    float s = 0.f;
    for (int k = t; k < DIN; k += 256) { float v = row[k]; s += v; orow[k] = (_Float16)v; }
    __shared__ float red[256];
    red[t] = s; __syncthreads();
    for (int o = 128; o > 0; o >>= 1) { if (t < o) red[t] += red[t + o]; __syncthreads(); }
    if (t == 0) {
        float m = (red[0] != 0.0f) ? 1.0f : 0.0f;
        maskf[r] = m;
        am_out[r] = 1.0f - m;   // attention_mask = ~mask
    }
}

// ---------------- generic weight transpose + f32->f16 pack: W[K x N] -> Wt[N x Kpad]
__global__ void k_packw(const float* __restrict__ W, _Float16* __restrict__ Wt,
                        int K, int N, int Kpad) {
    int idx = blockIdx.x * 256 + threadIdx.x;
    int total = N * Kpad;
    if (idx >= total) return;
    int n = idx / Kpad, k = idx % Kpad;
    float v = (k < K) ? W[(size_t)k * N + n] : 0.0f;
    Wt[idx] = (_Float16)v;
}

// ---------------- WMMA GEMM: C[M x N] = A_h[M x K] * Wt_h[N x K]^T --------------
// one wave = one 16x16 tile; K multiple of 32.
// flags: 1=relu, 2=*maskf[row], 4=*maskf[b,i]*maskf[b,j] (pair rows)
__global__ __launch_bounds__(32)
void k_gemm(const _Float16* __restrict__ A, int lda,
            const _Float16* __restrict__ B, int ldb,
            const float* __restrict__ bias,
            float* __restrict__ Cf, int ldc,
            _Float16* __restrict__ Ch, int ldch,
            int K, int flags, const float* __restrict__ maskf, int N2) {
    int mt = blockIdx.x, nt = blockIdx.y;
    int lane = threadIdx.x;
    int hf = lane >> 4, l = lane & 15;
    // A frag: row = mt*16+l ; halves at K = k0 + hf*8 + {0..7} and +16
    const _Float16* ap = A + (size_t)(mt * 16 + l) * lda + hf * 8;
    // B frag: col = nt*16+l ; halves at K = k0 + hf*8 + {0..7} and +16
    const _Float16* bp = B + (size_t)(nt * 16 + l) * ldb + hf * 8;
    v8f_t acc = {};
    for (int k0 = 0; k0 < K; k0 += 32) {
        v8h_t a0 = *(const v8h_t*)(ap + k0);
        v8h_t a1 = *(const v8h_t*)(ap + k0 + 16);
        v8h_t b0 = *(const v8h_t*)(bp + k0);
        v8h_t b1 = *(const v8h_t*)(bp + k0 + 16);
        v16h_t av = __builtin_shufflevector(a0, a1, 0,1,2,3,4,5,6,7,8,9,10,11,12,13,14,15);
        v16h_t bv = __builtin_shufflevector(b0, b1, 0,1,2,3,4,5,6,7,8,9,10,11,12,13,14,15);
        acc = __builtin_amdgcn_wmma_f32_16x16x32_f16(false, av, false, bv,
                                                     (short)0, acc, false, false);
    }
    int col = nt * 16 + l;
    float bc = bias ? bias[col] : 0.0f;
    #pragma unroll
    for (int v = 0; v < 8; ++v) {
        int row = mt * 16 + hf * 8 + v;           // C layout: lane=N, vgpr=M (+8 for hi half)
        float val = acc[v] + bc;
        if (flags & 1) val = fmaxf(val, 0.0f);
        if (flags & 2) val *= maskf[row];
        if (flags & 4) {
            int b = row / (N2 * N2), i = (row / N2) % N2, j = row % N2;
            val *= maskf[b * N2 + i] * maskf[b * N2 + j];
        }
        if (Cf) Cf[(size_t)row * ldc + col] = val;
        if (Ch) Ch[(size_t)row * ldch + col] = (_Float16)val;
    }
}

// ---------------- LayerNorm of y1 rows (512) -> d_out[:,0] f32 + xx_h f16 cols 0..511
__global__ void k_ln512(const float* __restrict__ y1, const float* __restrict__ g,
                        const float* __restrict__ be, float* __restrict__ outx,
                        _Float16* __restrict__ xx_h) {
    int r = blockIdx.x, t = threadIdx.x;
    const float* row = y1 + (size_t)r * DD;
    __shared__ float red[256];
    float s = 0.f;
    for (int k = t; k < DD; k += 256) s += row[k];
    red[t] = s; __syncthreads();
    for (int o = 128; o > 0; o >>= 1) { if (t < o) red[t] += red[t + o]; __syncthreads(); }
    float mean = red[0] / (float)DD; __syncthreads();
    float s2 = 0.f;
    for (int k = t; k < DD; k += 256) { float d = row[k] - mean; s2 += d * d; }
    red[t] = s2; __syncthreads();
    for (int o = 128; o > 0; o >>= 1) { if (t < o) red[t] += red[t + o]; __syncthreads(); }
    float rstd = rsqrtf(red[0] / (float)DD + 1e-5f);
    int b = r / NN, n = r % NN;
    float* od = outx + (((size_t)b * 2 + 0) * NN + n) * DD;
    _Float16* xh = xx_h + (size_t)r * DP;
    for (int k = t; k < DD; k += 256) {
        float v = (row[k] - mean) * rstd * g[k] + be[k];
        od[k] = v; xh[k] = (_Float16)v;
    }
    for (int k = 592 + t; k < DP; k += 256) xh[k] = (_Float16)0.0f;  // zero pad cols
}

// ---------------- pos MLP layer1 (BOX=8 -> 64) + ReLU + LN, f16 out --------------
__global__ void k_pos1(const float* __restrict__ box, const float* __restrict__ W,
                       const float* __restrict__ bi, const float* __restrict__ g,
                       const float* __restrict__ be, _Float16* __restrict__ out) {
    int r = blockIdx.x, t = threadIdx.x;  // 64 threads
    const float* br = box + (size_t)r * 8;
    float h = bi[t];
    #pragma unroll
    for (int k = 0; k < 8; ++k) h += br[k] * W[k * 64 + t];
    h = fmaxf(h, 0.0f);
    __shared__ float red[64];
    red[t] = h; __syncthreads();
    for (int o = 32; o > 0; o >>= 1) { if (t < o) red[t] += red[t + o]; __syncthreads(); }
    float mean = red[0] / 64.0f; __syncthreads();
    float d = h - mean;
    red[t] = d * d; __syncthreads();
    for (int o = 32; o > 0; o >>= 1) { if (t < o) red[t] += red[t + o]; __syncthreads(); }
    float rstd = rsqrtf(red[0] / 64.0f + 1e-5f);
    out[(size_t)r * 64 + t] = (_Float16)(d * rstd * g[t] + be[t]);
}

// ---------------- per-node rs_w dot products ------------------------------------
__global__ void k_rsdots(const _Float16* __restrict__ xs, const _Float16* __restrict__ xo,
                         const _Float16* __restrict__ ps, const float* __restrict__ rs_w,
                         float* __restrict__ subj_rs, float* __restrict__ obj_rs,
                         float* __restrict__ poss_rs) {
    int r = blockIdx.x, t = threadIdx.x;  // 128 threads
    float s1 = 0.f, s2 = 0.f, s3 = 0.f;
    for (int k = t; k < DD; k += 128) {
        float w = rs_w[k];
        s1 += (float)xs[(size_t)r * DD + k] * w;
        s2 += (float)xo[(size_t)r * DD + k] * w;
    }
    for (int k = t; k < 64; k += 128) s3 += (float)ps[(size_t)r * 64 + k] * rs_w[DD + k];
    __shared__ float ra[128], rb[128], rc[128];
    ra[t] = s1; rb[t] = s2; rc[t] = s3; __syncthreads();
    for (int o = 64; o > 0; o >>= 1) {
        if (t < o) { ra[t] += ra[t + o]; rb[t] += rb[t + o]; rc[t] += rc[t + o]; }
        __syncthreads();
    }
    if (t == 0) { subj_rs[r] = ra[0]; obj_rs[r] = rb[0]; poss_rs[r] = rc[0]; }
}

// ---------------- per-pair rs_w[512:576] dot ------------------------------------
__global__ void k_posr_rs(const _Float16* __restrict__ pr, const float* __restrict__ rs_w,
                          float* __restrict__ out, int R) {
    int r = blockIdx.x * 256 + threadIdx.x;
    if (r >= R) return;
    float s = 0.f;
    #pragma unroll 8
    for (int k = 0; k < 64; ++k) s += (float)pr[(size_t)r * 64 + k] * rs_w[DD + k];
    out[r] = s;
}

// ------- relationshipness: sigmoid p, masked softmax, attn-pooled relu feats ----
__global__ void k_relas(const float* __restrict__ subj_rs, const float* __restrict__ obj_rs,
                        const float* __restrict__ poss_rs, const float* __restrict__ posr_rs,
                        const float* __restrict__ subj_rf, const float* __restrict__ obj_rf,
                        const float* __restrict__ poss_rf, const _Float16* __restrict__ posr_rf,
                        const float* __restrict__ rs_b, const float* __restrict__ rf_b,
                        const float* __restrict__ maskf, float* __restrict__ p_out,
                        _Float16* __restrict__ xx_h) {
    int r = blockIdx.x;                    // (b,i)
    int b = r / NN, i = r % NN, t = threadIdx.x;  // 128 threads
    float mf_i = maskf[r];
    float rsb = rs_b[0];
    __shared__ float attn[NN];
    __shared__ float ssum;
    if (t < NN) {
        int j = t;
        float lg = (j == i) ? (subj_rs[r] + poss_rs[r] + rsb)
                            : (mf_i * obj_rs[b * NN + j] + posr_rs[(size_t)r * NN + j] + rsb);
        float pm = mf_i * maskf[b * NN + j];
        p_out[(size_t)r * NN + j] = pm * (1.0f / (1.0f + __expf(-lg)));
        attn[j] = (pm > 0.0f) ? lg : -1e9f;
    }
    __syncthreads();
    if (t == 0) {
        float m = -3.0e38f;
        for (int j = 0; j < NN; ++j) m = fmaxf(m, attn[j]);
        float s = 0.f;
        for (int j = 0; j < NN; ++j) { float e = __expf(attn[j] - m); attn[j] = e; s += e; }
        ssum = s;
    }
    __syncthreads();
    float inv = 1.0f / ssum;
    for (int f = t; f < 80; f += 128) {
        float fb = rf_b[f], acc = 0.f;
        for (int j = 0; j < NN; ++j) {
            float pre = (j == i)
                ? (subj_rf[(size_t)r * 80 + f] + poss_rf[(size_t)r * 80 + f] + fb)
                : (mf_i * obj_rf[(size_t)(b * NN + j) * 80 + f]
                   + (float)posr_rf[((size_t)r * NN + j) * 80 + f] + fb);
            acc += attn[j] * fmaxf(pre, 0.0f);
        }
        xx_h[(size_t)r * DP + DD + f] = (_Float16)(acc * inv * mf_i);
    }
}

// ---------------- degree rsqrt of thresholded adjacency -------------------------
__global__ void k_dinv(const float* __restrict__ p, float* __restrict__ dinv) {
    int x = blockIdx.x * 256 + threadIdx.x;
    if (x >= M1) return;
    int b = x / NN, j = x % NN;
    float deg = 0.f;
    for (int i2 = 0; i2 < NN; ++i2) {
        float a = p[((size_t)b * NN + i2) * NN + j];
        deg += ((a > THRV) || (i2 == j)) ? 1.0f : 0.0f;
    }
    dinv[x] = rsqrtf(deg);
}

// ---------------- GCN aggregation: out[b,j,f] = relu(sum_i An[b,i,j]*t[b,i,f]+bias)
__global__ void k_gcn_agg(const float* __restrict__ t_in, int W,
                          const float* __restrict__ bias, const float* __restrict__ p,
                          const float* __restrict__ dinv,
                          _Float16* __restrict__ out_h, int ldo, int zeroPadTo,
                          float* __restrict__ out_f) {
    int r = blockIdx.x;                  // (b,j)
    int b = r / NN, j = r % NN, t = threadIdx.x;  // 128 threads
    __shared__ float an[NN];
    if (t < NN) {
        int i2 = t;
        float pv = p[((size_t)b * NN + i2) * NN + j];
        bool e = (pv > THRV) || (i2 == j);
        an[i2] = e ? dinv[b * NN + i2] * dinv[b * NN + j] : 0.0f;
    }
    __syncthreads();
    for (int f = t; f < W; f += 128) {
        float acc = bias[f];
        for (int i2 = 0; i2 < NN; ++i2) acc += an[i2] * t_in[((size_t)b * NN + i2) * W + f];
        acc = fmaxf(acc, 0.0f);
        if (out_h) out_h[(size_t)r * ldo + f] = (_Float16)acc;
        if (out_f) out_f[(((size_t)b * 2 + 1) * NN + j) * DD + f] = acc;
    }
    if (out_h)
        for (int f = W + t; f < zeroPadTo; f += 128) out_h[(size_t)r * ldo + f] = (_Float16)0.0f;
}

extern "C" void kernel_launch(void* const* d_in, const int* in_sizes, int n_in,
                              void* d_out, int out_size, void* d_ws, size_t ws_size,
                              hipStream_t stream) {
    const float* images   = (const float*)d_in[0];
    const float* selfbbox = (const float*)d_in[1];
    const float* bbox     = (const float*)d_in[2];
    const float* pre_w = (const float*)d_in[3];  const float* pre_b = (const float*)d_in[4];
    const float* pre_g = (const float*)d_in[5];  const float* pre_be = (const float*)d_in[6];
    const float* subj_w = (const float*)d_in[7]; const float* subj_b = (const float*)d_in[8];
    const float* obj_w = (const float*)d_in[9];  const float* obj_b = (const float*)d_in[10];
    const float* ps1_w = (const float*)d_in[11]; const float* ps1_b = (const float*)d_in[12];
    const float* ps_g = (const float*)d_in[13];  const float* ps_be = (const float*)d_in[14];
    const float* ps2_w = (const float*)d_in[15]; const float* ps2_b = (const float*)d_in[16];
    const float* pr1_w = (const float*)d_in[17]; const float* pr1_b = (const float*)d_in[18];
    const float* pr_g = (const float*)d_in[19];  const float* pr_be = (const float*)d_in[20];
    const float* pr2_w = (const float*)d_in[21]; const float* pr2_b = (const float*)d_in[22];
    const float* rs_w = (const float*)d_in[23];  const float* rs_b = (const float*)d_in[24];
    const float* rf_w = (const float*)d_in[25];  const float* rf_b = (const float*)d_in[26];
    const float* g1_w = (const float*)d_in[27];  const float* g1_b = (const float*)d_in[28];
    const float* g2_w = (const float*)d_in[29];  const float* g2_b = (const float*)d_in[30];
    const float* g3_w = (const float*)d_in[31];  const float* g3_b = (const float*)d_in[32];

    float* outf = (float*)d_out;
    float* am_out = outf + (size_t)BV * 2 * NN * DD;   // attention mask tail

    // ---- workspace allocator (256B aligned) ----
    char* wp = (char*)d_ws;
    auto alloc = [&](size_t bytes) -> void* {
        void* p = (void*)wp;
        wp += (bytes + 255) & ~(size_t)255;
        return p;
    };
    _Float16* A_img    = ( _Float16*)alloc((size_t)M1 * DIN * 2);
    float*    maskf    = (float*)    alloc((size_t)M1 * 4);
    _Float16* preT     = ( _Float16*)alloc((size_t)DD * DIN * 2);
    _Float16* subjT    = ( _Float16*)alloc((size_t)DD * DD * 2);
    _Float16* objT     = ( _Float16*)alloc((size_t)DD * DD * 2);
    _Float16* ps2T     = ( _Float16*)alloc((size_t)64 * 64 * 2);
    _Float16* pr2T     = ( _Float16*)alloc((size_t)64 * 64 * 2);
    _Float16* rfT      = ( _Float16*)alloc((size_t)80 * 576 * 2);
    _Float16* g1T      = ( _Float16*)alloc((size_t)592 * DP * 2);
    _Float16* g2T      = ( _Float16*)alloc((size_t)DD * DP * 2);
    _Float16* g3T      = ( _Float16*)alloc((size_t)DD * DD * 2);
    float*    y1       = (float*)    alloc((size_t)M1 * DD * 4);
    _Float16* xx_h     = ( _Float16*)alloc((size_t)M1 * DP * 2);
    _Float16* xsubj_h  = ( _Float16*)alloc((size_t)M1 * DD * 2);
    _Float16* xobj_h   = ( _Float16*)alloc((size_t)M1 * DD * 2);
    _Float16* hself_h  = ( _Float16*)alloc((size_t)M1 * 64 * 2);
    _Float16* posself_h= ( _Float16*)alloc((size_t)M1 * 64 * 2);
    _Float16* hrel_h   = ( _Float16*)alloc((size_t)PP * 64 * 2);
    _Float16* posrel_h = ( _Float16*)alloc((size_t)PP * 64 * 2);
    float*    obj_rf   = (float*)    alloc((size_t)M1 * 80 * 4);
    float*    subj_rf  = (float*)    alloc((size_t)M1 * 80 * 4);
    float*    poss_rf  = (float*)    alloc((size_t)M1 * 80 * 4);
    _Float16* posr_rf  = ( _Float16*)alloc((size_t)PP * 80 * 2);
    float*    subj_rs  = (float*)    alloc((size_t)M1 * 4);
    float*    obj_rs   = (float*)    alloc((size_t)M1 * 4);
    float*    poss_rs  = (float*)    alloc((size_t)M1 * 4);
    float*    posr_rs  = (float*)    alloc((size_t)PP * 4);
    float*    p_buf    = (float*)    alloc((size_t)PP * 4);
    float*    dinv     = (float*)    alloc((size_t)M1 * 4);
    float*    t_buf    = (float*)    alloc((size_t)M1 * 592 * 4);
    _Float16* x1_h     = ( _Float16*)alloc((size_t)M1 * DP * 2);
    _Float16* x2_h     = ( _Float16*)alloc((size_t)M1 * DD * 2);

    // ---- stage 0: mask + f16 images; pack all weights transposed/padded ----
    k_mask_img<<<M1, 256, 0, stream>>>(images, A_img, maskf, am_out);
    auto packw = [&](const float* W, _Float16* Wt, int K, int N, int Kpad) {
        int total = N * Kpad;
        k_packw<<<(total + 255) / 256, 256, 0, stream>>>(W, Wt, K, N, Kpad);
    };
    packw(pre_w, preT, DIN, DD, DIN);
    packw(subj_w, subjT, DD, DD, DD);
    packw(obj_w, objT, DD, DD, DD);
    packw(ps2_w, ps2T, 64, 64, 64);
    packw(pr2_w, pr2T, 64, 64, 64);
    packw(rf_w, rfT, 576, 80, 576);
    packw(g1_w, g1T, 592, 592, DP);
    packw(g2_w, g2T, 592, DD, DP);
    packw(g3_w, g3T, DD, DD, DD);

    // ---- preprocess: y1 = relu(images@pre_w + pre_b); x = LN(y1) ----
    k_gemm<<<dim3(M1/16, DD/16), 32, 0, stream>>>(A_img, DIN, preT, DIN, pre_b,
        y1, DD, (_Float16*)nullptr, 0, DIN, /*relu*/1, nullptr, NN);
    k_ln512<<<M1, 256, 0, stream>>>(y1, pre_g, pre_be, outf, xx_h);

    // ---- x_subj / x_obj (f16 out, masked) ----
    k_gemm<<<dim3(M1/16, DD/16), 32, 0, stream>>>(xx_h, DP, subjT, DD, subj_b,
        (float*)nullptr, 0, xsubj_h, DD, DD, 1|2, maskf, NN);
    k_gemm<<<dim3(M1/16, DD/16), 32, 0, stream>>>(xx_h, DP, objT, DD, obj_b,
        (float*)nullptr, 0, xobj_h, DD, DD, 1|2, maskf, NN);

    // ---- positional MLPs ----
    k_pos1<<<M1, 64, 0, stream>>>(selfbbox, ps1_w, ps1_b, ps_g, ps_be, hself_h);
    k_pos1<<<PP, 64, 0, stream>>>(bbox, pr1_w, pr1_b, pr_g, pr_be, hrel_h);
    k_gemm<<<dim3(M1/16, 64/16), 32, 0, stream>>>(hself_h, 64, ps2T, 64, ps2_b,
        (float*)nullptr, 0, posself_h, 64, 64, 1|2, maskf, NN);
    k_gemm<<<dim3(PP/16, 64/16), 32, 0, stream>>>(hrel_h, 64, pr2T, 64, pr2_b,
        (float*)nullptr, 0, posrel_h, 64, 64, 1|4, maskf, NN);

    // ---- factored pair projections through rf_w ----
    k_gemm<<<dim3(M1/16, 80/16), 32, 0, stream>>>(xobj_h, DD, rfT, 576, nullptr,
        obj_rf, 80, (_Float16*)nullptr, 0, DD, 0, nullptr, NN);
    k_gemm<<<dim3(M1/16, 80/16), 32, 0, stream>>>(xsubj_h, DD, rfT, 576, nullptr,
        subj_rf, 80, (_Float16*)nullptr, 0, DD, 0, nullptr, NN);
    k_gemm<<<dim3(M1/16, 80/16), 32, 0, stream>>>(posself_h, 64, rfT + 512, 576, nullptr,
        poss_rf, 80, (_Float16*)nullptr, 0, 64, 0, nullptr, NN);
    k_gemm<<<dim3(PP/16, 80/16), 32, 0, stream>>>(posrel_h, 64, rfT + 512, 576, nullptr,
        (float*)nullptr, 0, posr_rf, 80, 64, 0, nullptr, NN);

    // ---- factored rs_w dot products ----
    k_rsdots<<<M1, 128, 0, stream>>>(xsubj_h, xobj_h, posself_h, rs_w,
                                     subj_rs, obj_rs, poss_rs);
    k_posr_rs<<<(PP + 255) / 256, 256, 0, stream>>>(posrel_h, rs_w, posr_rs, PP);

    // ---- sigmoid p + masked softmax + attn-pooled feats -> xx cols 512..591 ----
    k_relas<<<M1, 128, 0, stream>>>(subj_rs, obj_rs, poss_rs, posr_rs,
                                    subj_rf, obj_rf, poss_rf, posr_rf,
                                    rs_b, rf_b, maskf, p_buf, xx_h);

    // ---- adjacency normalization ----
    k_dinv<<<(M1 + 255) / 256, 256, 0, stream>>>(p_buf, dinv);

    // ---- GCN layer 1 ----
    k_gemm<<<dim3(M1/16, 592/16), 32, 0, stream>>>(xx_h, DP, g1T, DP, nullptr,
        t_buf, 592, (_Float16*)nullptr, 0, DP, 0, nullptr, NN);
    k_gcn_agg<<<M1, 128, 0, stream>>>(t_buf, 592, g1_b, p_buf, dinv, x1_h, DP, DP, nullptr);
    // ---- GCN layer 2 ----
    k_gemm<<<dim3(M1/16, DD/16), 32, 0, stream>>>(x1_h, DP, g2T, DP, nullptr,
        t_buf, DD, (_Float16*)nullptr, 0, DP, 0, nullptr, NN);
    k_gcn_agg<<<M1, 128, 0, stream>>>(t_buf, DD, g2_b, p_buf, dinv, x2_h, DD, DD, nullptr);
    // ---- GCN layer 3 -> outs[:,1] ----
    k_gemm<<<dim3(M1/16, DD/16), 32, 0, stream>>>(x2_h, DD, g3T, DD, nullptr,
        t_buf, DD, (_Float16*)nullptr, 0, DD, 0, nullptr, NN);
    k_gcn_agg<<<M1, 128, 0, stream>>>(t_buf, DD, g3_b, p_buf, dinv,
        (_Float16*)nullptr, 0, 0, outf);

    (void)in_sizes; (void)n_in; (void)out_size; (void)ws_size;
}
